// VirtualFourier_83133386981930
// MI455X (gfx1250) — compile-verified, matrix operand order checked by reading, not run on hardware
//
#include <hip/hip_runtime.h>

// ---------------------------------------------------------------------------
// VirtualFourier on gfx1250 (MI455X): bf16 WMMA GEMM pipeline (128x64 block
// tiles, double-buffered LDS, global_prefetch), softmax fused into GEMM LDS
// staging, truncated-DFT Fourier mixing kernel.
// B=4 T=4096 C=1024 H=16 S=256 M=32 D=64
// ---------------------------------------------------------------------------

typedef __attribute__((ext_vector_type(16))) __bf16 v16bf;
typedef __attribute__((ext_vector_type(8)))  float  v8f;

#define LSTRIDE     36   // padded element stride for [row][K=32] LDS tiles
#define LSTRIDE_KN  68   // padded element stride for [K=32][N=64] LDS tiles
#define ATILE      (128 * LSTRIDE)     // elements per A double-buffer slot
#define BTILE      (64 * LSTRIDE)      // elements per B (N-major) slot
#define BTILE_KN   (32 * LSTRIDE_KN)   // elements per B (K-major) slot

// A/B fragment from M-major (or N-major) LDS tile [row][k], row=lane&15.
// 16-bit layout (ISA 7.12.2): VGPR j holds a K pair; lanes>=16 shifted +8/+16.
__device__ __forceinline__ v16bf frag_ld_mk(const __bf16* t, int lane) {
  const int row = lane & 15;
  const int kh  = (lane >> 4) << 3;
  const __bf16* r = t + row * LSTRIDE;
  v16bf f;
#pragma unroll
  for (int j = 0; j < 8; ++j) {
    const int kb = ((j >> 2) << 4) + kh + ((j & 3) << 1);
    f[2 * j]     = r[kb];
    f[2 * j + 1] = r[kb + 1];
  }
  return f;
}

// B fragment from K-major LDS tile [k][n], n=lane&15.
__device__ __forceinline__ v16bf frag_ld_kn(const __bf16* t, int lane) {
  const int n  = lane & 15;
  const int kh = (lane >> 4) << 3;
  v16bf f;
#pragma unroll
  for (int j = 0; j < 8; ++j) {
    const int kb = ((j >> 2) << 4) + kh + ((j & 3) << 1);
    f[2 * j]     = t[kb * LSTRIDE_KN + n];
    f[2 * j + 1] = t[(kb + 1) * LSTRIDE_KN + n];
  }
  return f;
}

__device__ __forceinline__ v8f wmma_bf16(v16bf a, v16bf b, v8f c) {
  return __builtin_amdgcn_wmma_f32_16x16x32_bf16(false, a, false, b,
                                                 (short)0, c, false, false);
}

// ---------------------------------------------------------------------------
__global__ __launch_bounds__(256) void f2bf_kernel(const float* __restrict__ in,
                                                   __bf16* __restrict__ out, int n) {
  const int i = blockIdx.x * 256 + threadIdx.x;
  if (i < n) out[i] = (__bf16)in[i];
}

// ---------------------------------------------------------------------------
// Large GEMM: C[M x N] = A[M x K](bf16,row-major) * Bt[N x K](bf16)^T + bias
// 128x64 block tile, 8 waves, each wave: 4 M-tiles x 1 N-tile, double-buffered.
// mode 0: in_proj -> scatter bf16 into tmp[(b,h,t,d)]
// mode 1: out_proj -> fp32 row-major output
__global__ __launch_bounds__(256) void gemm_big_kernel(
    const __bf16* __restrict__ A, const __bf16* __restrict__ Bt,
    const float* __restrict__ bias, float* __restrict__ outF,
    __bf16* __restrict__ outScatter, int K, int lda, int ldb, int N, int mode)
{
  __shared__ __bf16 As[2 * ATILE];
  __shared__ __bf16 Bs[2 * BTILE];
  const int tid = threadIdx.x, lane = tid & 31, wave = tid >> 5;
  const int m0 = blockIdx.x * 128, n0 = blockIdx.y * 64;
  const int mtb = (wave >> 2) << 6, nt = (wave & 3) << 4;
  const int nk = K >> 5;

  auto stage = [&](int kt, int buf) {
    const int k0 = kt << 5;
    __bf16* as = As + buf * ATILE;
    __bf16* bs = Bs + buf * BTILE;
#pragma unroll
    for (int c = 0; c < 4; ++c) {
      const int idx = tid + (c << 8);
      const int r = idx >> 3, cc = (idx & 7) << 2;
      *(uint2*)&as[r * LSTRIDE + cc] = *(const uint2*)&A[(size_t)(m0 + r) * lda + k0 + cc];
    }
#pragma unroll
    for (int c = 0; c < 2; ++c) {
      const int idx = tid + (c << 8);
      const int r = idx >> 3, cc = (idx & 7) << 2;
      *(uint2*)&bs[r * LSTRIDE + cc] = *(const uint2*)&Bt[(size_t)(n0 + r) * ldb + k0 + cc];
    }
  };

  const v8f zero = {};
  v8f acc[4] = {zero, zero, zero, zero};
  stage(0, 0);
  __syncthreads();
  for (int k = 0; k < nk; ++k) {
    const int cur = k & 1;
    if (k + 1 < nk) stage(k + 1, cur ^ 1);
    if (k + 2 < nk) {
      const int kp = (k + 2) << 5;
      if (tid < 128) __builtin_prefetch(&A[(size_t)(m0 + tid) * lda + kp], 0, 1);
      else           __builtin_prefetch(&Bt[(size_t)(n0 + (tid & 63)) * ldb + kp], 0, 1);
    }
    const __bf16* as = As + cur * ATILE;
    const __bf16* bs = Bs + cur * BTILE;
    const v16bf b = frag_ld_mk(bs + nt * LSTRIDE, lane);
#pragma unroll
    for (int i = 0; i < 4; ++i) {
      const v16bf a = frag_ld_mk(as + (mtb + (i << 4)) * LSTRIDE, lane);
      acc[i] = wmma_bf16(a, b, acc[i]);
    }
    __syncthreads();
  }

  const int col = n0 + nt + (lane & 15);
  const int rbase = (lane >> 4) << 3;
  const float bv = bias[col];
  if (mode == 0) {
    const int h = col >> 6, d = col & 63;
#pragma unroll
    for (int i = 0; i < 4; ++i) {
#pragma unroll
      for (int r = 0; r < 8; ++r) {
        const int gm = m0 + mtb + (i << 4) + rbase + r;
        const int bb = gm >> 12, t = gm & 4095;
        outScatter[((size_t)(bb * 16 + h) * 4096 + t) * 64 + d] = (__bf16)(acc[i][r] + bv);
      }
    }
  } else {
#pragma unroll
    for (int i = 0; i < 4; ++i) {
#pragma unroll
      for (int r = 0; r < 8; ++r) {
        const int gm = m0 + mtb + (i << 4) + rbase + r;
        outF[(size_t)gm * N + col] = acc[i][r] + bv;
      }
    }
  }
}

// ---------------------------------------------------------------------------
// logits[bh][t][s] = (tmp[bh] @ Ws^T + bs) / temp[h]    (M=4096 N=256 K=64)
__global__ __launch_bounds__(256) void gemm_logits_kernel(
    const __bf16* __restrict__ tmp, const __bf16* __restrict__ Ws,
    const float* __restrict__ bs, const float* __restrict__ temp,
    float* __restrict__ logits)
{
  __shared__ __bf16 As[2 * ATILE];
  __shared__ __bf16 Bs[2 * BTILE];
  const int tid = threadIdx.x, lane = tid & 31, wave = tid >> 5;
  const int bh = blockIdx.z;
  const int m0 = blockIdx.x * 128, n0 = blockIdx.y * 64;
  const int mtb = (wave >> 2) << 6, nt = (wave & 3) << 4;
  const __bf16* A = tmp + (size_t)bh * 4096 * 64;

  auto stage = [&](int kt, int buf) {
    const int k0 = kt << 5;
    __bf16* as = As + buf * ATILE;
    __bf16* bs2 = Bs + buf * BTILE;
#pragma unroll
    for (int c = 0; c < 4; ++c) {
      const int idx = tid + (c << 8);
      const int r = idx >> 3, cc = (idx & 7) << 2;
      *(uint2*)&as[r * LSTRIDE + cc] = *(const uint2*)&A[(size_t)(m0 + r) * 64 + k0 + cc];
    }
#pragma unroll
    for (int c = 0; c < 2; ++c) {
      const int idx = tid + (c << 8);
      const int r = idx >> 3, cc = (idx & 7) << 2;
      *(uint2*)&bs2[r * LSTRIDE + cc] = *(const uint2*)&Ws[(size_t)(n0 + r) * 64 + k0 + cc];
    }
  };

  const v8f zero = {};
  v8f acc[4] = {zero, zero, zero, zero};
  stage(0, 0);
  __syncthreads();
  for (int k = 0; k < 2; ++k) {
    const int cur = k & 1;
    if (k + 1 < 2) stage(k + 1, cur ^ 1);
    const __bf16* as = As + cur * ATILE;
    const __bf16* bs2 = Bs + cur * BTILE;
    const v16bf b = frag_ld_mk(bs2 + nt * LSTRIDE, lane);
#pragma unroll
    for (int i = 0; i < 4; ++i) {
      const v16bf a = frag_ld_mk(as + (mtb + (i << 4)) * LSTRIDE, lane);
      acc[i] = wmma_bf16(a, b, acc[i]);
    }
    __syncthreads();
  }

  const float invt = 1.0f / temp[bh & 15];
  const int col = n0 + nt + (lane & 15);
  const int rbase = (lane >> 4) << 3;
  const float bv = bs[col];
  float* lg = logits + (size_t)bh * 4096 * 256;
#pragma unroll
  for (int i = 0; i < 4; ++i) {
#pragma unroll
    for (int r = 0; r < 8; ++r) {
      const int gm = m0 + mtb + (i << 4) + rbase + r;
      lg[(size_t)gm * 256 + col] = (acc[i][r] + bv) * invt;
    }
  }
}

// ---------------------------------------------------------------------------
// Per-row (over S=256) softmax stats: one wave32 per row, 8 rows per block.
__global__ __launch_bounds__(256) void rowstats_kernel(const float* __restrict__ logits,
                                                       float* __restrict__ rmax,
                                                       float* __restrict__ rsum)
{
  const int row = blockIdx.x * 8 + (threadIdx.x >> 5);
  const int lane = threadIdx.x & 31;
  const float* r = logits + (size_t)row * 256;
  float v[8];
  float mx = -3.0e38f;
#pragma unroll
  for (int j = 0; j < 8; ++j) { v[j] = r[lane + (j << 5)]; mx = fmaxf(mx, v[j]); }
#pragma unroll
  for (int off = 16; off > 0; off >>= 1) mx = fmaxf(mx, __shfl_xor(mx, off, 32));
  float s = 0.0f;
#pragma unroll
  for (int j = 0; j < 8; ++j) s += __expf(v[j] - mx);
#pragma unroll
  for (int off = 16; off > 0; off >>= 1) s += __shfl_xor(s, off, 32);
  if (lane == 0) { rmax[row] = mx; rsum[row] = s; }
}

// Per-column (over T=4096) softmax stats: one block per (b,h), thread = s.
__global__ __launch_bounds__(256) void colstats_kernel(const float* __restrict__ logits,
                                                       float* __restrict__ cmax,
                                                       float* __restrict__ csum)
{
  const int bh = blockIdx.x, s = threadIdx.x;
  const float* base = logits + (size_t)bh * 4096 * 256 + s;
  float mx = -3.0e38f;
  for (int t = 0; t < 4096; ++t) mx = fmaxf(mx, base[(size_t)t * 256]);
  float sm = 0.0f;
  for (int t = 0; t < 4096; ++t) sm += __expf(base[(size_t)t * 256] - mx);
  cmax[bh * 256 + s] = mx;
  csum[bh * 256 + s] = sm;
}

// ---------------------------------------------------------------------------
// z[bh][s][d] = sum_t softmax_t(logits)[t][s] * tmp[bh][t][d]
// M=256(s) N=64(d) K=4096(t); column softmax applied during transposed staging.
__global__ __launch_bounds__(256) void gemm_pool_kernel(
    const float* __restrict__ logits, const __bf16* __restrict__ tmp,
    const float* __restrict__ cmax, const float* __restrict__ csum,
    float* __restrict__ z)
{
  __shared__ __bf16 As[2 * ATILE];      // [s:128][t:32]
  __shared__ __bf16 Bs[2 * BTILE_KN];   // [t:32][d:64] K-major
  const int tid = threadIdx.x, lane = tid & 31, wave = tid >> 5;
  const int bh = blockIdx.z;
  const int m0 = blockIdx.x * 128;
  const int mtb = (wave >> 2) << 6, nt = (wave & 3) << 4;
  const float* lg = logits + (size_t)bh * 4096 * 256;
  const int tr = tid >> 3;        // t row inside k-tile (0..31)
  const int sc = (tid & 7) << 3;  // s offset within 64-group
  float cm[16], ci[16];
#pragma unroll
  for (int g = 0; g < 2; ++g) {
#pragma unroll
    for (int j = 0; j < 8; ++j) {
      const int sg = m0 + (g << 6) + sc + j;
      cm[g * 8 + j] = cmax[bh * 256 + sg];
      ci[g * 8 + j] = 1.0f / csum[bh * 256 + sg];
    }
  }

  auto stage = [&](int kt, int buf) {
    const int k0 = kt << 5;
    __bf16* as = As + buf * ATILE;
    __bf16* bs = Bs + buf * BTILE_KN;
#pragma unroll
    for (int g = 0; g < 2; ++g) {
#pragma unroll
      for (int j = 0; j < 8; ++j) {   // transpose + column softmax
        const float v = lg[(size_t)(k0 + tr) * 256 + m0 + (g << 6) + sc + j];
        as[((g << 6) + sc + j) * LSTRIDE + tr] =
            (__bf16)(__expf(v - cm[g * 8 + j]) * ci[g * 8 + j]);
      }
    }
#pragma unroll
    for (int c = 0; c < 2; ++c) {     // tmp tile, K-major copy
      const int idx = tid + (c << 8);
      const int r = idx >> 4, cc = (idx & 15) << 2;
      *(uint2*)&bs[r * LSTRIDE_KN + cc] =
          *(const uint2*)&tmp[((size_t)bh * 4096 + k0 + r) * 64 + cc];
    }
  };

  const v8f zero = {};
  v8f acc[4] = {zero, zero, zero, zero};
  const int nk = 128;  // 4096/32
  stage(0, 0);
  __syncthreads();
  for (int k = 0; k < nk; ++k) {
    const int cur = k & 1;
    if (k + 1 < nk) stage(k + 1, cur ^ 1);
    if (k + 2 < nk) {   // prefetch next-next logits tile rows
      const int kp = (k + 2) << 5;
      const int pr = tid >> 3, pc = (tid & 7) << 4;
      __builtin_prefetch(&lg[(size_t)(kp + pr) * 256 + m0 + pc], 0, 1);
    }
    const __bf16* as = As + cur * ATILE;
    const __bf16* bs = Bs + cur * BTILE_KN;
    const v16bf b = frag_ld_kn(bs + nt, lane);
#pragma unroll
    for (int i = 0; i < 4; ++i) {
      const v16bf a = frag_ld_mk(as + (mtb + (i << 4)) * LSTRIDE, lane);
      acc[i] = wmma_bf16(a, b, acc[i]);
    }
    __syncthreads();
  }

  const int col = nt + (lane & 15);
  const int rbase = (lane >> 4) << 3;
  float* zb = z + (size_t)bh * 256 * 64;
#pragma unroll
  for (int i = 0; i < 4; ++i) {
#pragma unroll
    for (int r = 0; r < 8; ++r) {
      const int gm = m0 + mtb + (i << 4) + rbase + r;
      zb[(size_t)gm * 64 + col] = acc[i][r];
    }
  }
}

// ---------------------------------------------------------------------------
// Truncated rfft(S=256 -> M=32 modes) + complex D x D mode mixing + irfft.
// One block per (b,h); writes outT[bh][d][s] bf16 (N-major B for scatter GEMM).
__global__ __launch_bounds__(256) void fourier_kernel(
    const float* __restrict__ z, const float* __restrict__ w_re,
    const float* __restrict__ w_im, __bf16* __restrict__ outT)
{
  __shared__ float Zre[32 * 64], Zim[32 * 64], Ore[32 * 64], Oim[32 * 64];
  const int bh = blockIdx.x, h = bh & 15;
  const int tid = threadIdx.x;
  const float* zb = z + (size_t)bh * 256 * 64;
  const float TWO_PI = 6.283185307179586f;

  {   // forward DFT: thread owns i = tid&63, modes m = (tid>>6) + 4*jj
    const int i = tid & 63, mq = tid >> 6;
    float are[8] = {0}, aim[8] = {0};
    for (int s = 0; s < 256; ++s) {
      const float zv = zb[s * 64 + i];
#pragma unroll
      for (int jj = 0; jj < 8; ++jj) {
        const int m = mq + 4 * jj;
        float c, sn;
        __sincosf(-TWO_PI * (float)(m * s) * (1.0f / 256.0f), &sn, &c);
        are[jj] += zv * c;
        aim[jj] += zv * sn;
      }
    }
#pragma unroll
    for (int jj = 0; jj < 8; ++jj) {
      Zre[(mq + 4 * jj) * 64 + i] = are[jj];
      Zim[(mq + 4 * jj) * 64 + i] = aim[jj];
    }
  }
  __syncthreads();
  {   // mode mixing: O[m][o] = sum_i Z[m][i] * (wre + i*wim)[h][i][o][m]
    const int o = tid & 63, mq = tid >> 6;
    float ore[8] = {0}, oim[8] = {0};
    for (int i = 0; i < 64; ++i) {
#pragma unroll
      for (int jj = 0; jj < 8; ++jj) {
        const int m = mq + 4 * jj;
        const float zr = Zre[m * 64 + i], zi = Zim[m * 64 + i];
        const size_t widx = (((size_t)(h * 64 + i) * 64 + o) * 32) + m;
        const float wr = w_re[widx], wi = w_im[widx];
        ore[jj] += zr * wr - zi * wi;
        oim[jj] += zr * wi + zi * wr;
      }
    }
    __syncthreads();
#pragma unroll
    for (int jj = 0; jj < 8; ++jj) {
      Ore[(mq + 4 * jj) * 64 + o] = ore[jj];
      Oim[(mq + 4 * jj) * 64 + o] = oim[jj];
    }
  }
  __syncthreads();
  {   // irfft (only 32 live modes) + transposed store [d][s]
    const int o = tid & 63, sq = tid >> 6;
    __bf16* ob = outT + (size_t)bh * 64 * 256;
    for (int ss = 0; ss < 64; ++ss) {
      const int s = sq * 64 + ss;
      float val = Ore[o];  // m=0 (imag ignored, per irfft semantics)
#pragma unroll
      for (int m = 1; m < 32; ++m) {
        float c, sn;
        __sincosf(TWO_PI * (float)(m * s) * (1.0f / 256.0f), &sn, &c);
        val += 2.0f * (Ore[m * 64 + o] * c - Oim[m * 64 + o] * sn);
      }
      ob[o * 256 + s] = (__bf16)(val * (1.0f / 256.0f));
    }
  }
}

// ---------------------------------------------------------------------------
// y2[b][t][h*64+d] = sum_s softmax_s(logits)[t][s] * outPool[s][d]
// M=4096(t) N=64(d) K=256(s); row softmax applied during A staging.
__global__ __launch_bounds__(256) void gemm_scatter_kernel(
    const float* __restrict__ logits, const __bf16* __restrict__ outT,
    const float* __restrict__ rmax, const float* __restrict__ rsum,
    __bf16* __restrict__ y2)
{
  __shared__ __bf16 As[2 * ATILE];   // [t:128][s:32]
  __shared__ __bf16 Bs[2 * BTILE];   // [d:64][s:32]
  const int tid = threadIdx.x, lane = tid & 31, wave = tid >> 5;
  const int bh = blockIdx.z;
  const int m0 = blockIdx.x * 128;
  const int mtb = (wave >> 2) << 6, nt = (wave & 3) << 4;
  const float* lg = logits + (size_t)bh * 4096 * 256;
  const int rowl = tid >> 1;          // 0..127
  const int cbase = (tid & 1) << 4;   // 0 or 16
  const float rm = rmax[bh * 4096 + m0 + rowl];
  const float ri = 1.0f / rsum[bh * 4096 + m0 + rowl];

  auto stage = [&](int kt, int buf) {
    const int k0 = kt << 5;
    __bf16* as = As + buf * ATILE;
    __bf16* bs = Bs + buf * BTILE;
#pragma unroll
    for (int j = 0; j < 16; ++j) {   // row softmax during staging
      const float v = lg[(size_t)(m0 + rowl) * 256 + k0 + cbase + j];
      as[rowl * LSTRIDE + cbase + j] = (__bf16)(__expf(v - rm) * ri);
    }
#pragma unroll
    for (int c = 0; c < 2; ++c) {
      const int idx = tid + (c << 8);
      const int r = idx >> 3, cw = (idx & 7) << 2;
      *(uint2*)&bs[r * LSTRIDE + cw] =
          *(const uint2*)&outT[((size_t)bh * 64 + r) * 256 + k0 + cw];
    }
  };

  const v8f zero = {};
  v8f acc[4] = {zero, zero, zero, zero};
  const int nk = 8;  // 256/32
  stage(0, 0);
  __syncthreads();
  for (int k = 0; k < nk; ++k) {
    const int cur = k & 1;
    if (k + 1 < nk) stage(k + 1, cur ^ 1);
    const __bf16* as = As + cur * ATILE;
    const __bf16* bs = Bs + cur * BTILE;
    const v16bf b = frag_ld_mk(bs + nt * LSTRIDE, lane);
#pragma unroll
    for (int i = 0; i < 4; ++i) {
      const v16bf a = frag_ld_mk(as + (mtb + (i << 4)) * LSTRIDE, lane);
      acc[i] = wmma_bf16(a, b, acc[i]);
    }
    __syncthreads();
  }

  const int b = bh >> 4, h = bh & 15;
  const int col = h * 64 + nt + (lane & 15);
  const int rbase = (lane >> 4) << 3;
#pragma unroll
  for (int i = 0; i < 4; ++i) {
#pragma unroll
    for (int r = 0; r < 8; ++r) {
      const int t = m0 + mtb + (i << 4) + rbase + r;
      y2[((size_t)b * 4096 + t) * 1024 + col] = (__bf16)acc[i][r];
    }
  }
}

// ---------------------------------------------------------------------------
extern "C" void kernel_launch(void* const* d_in, const int* in_sizes, int n_in,
                              void* d_out, int out_size, void* d_ws, size_t ws_size,
                              hipStream_t stream) {
  (void)in_sizes; (void)n_in; (void)out_size; (void)ws_size;
  const float* x_q  = (const float*)d_in[0];
  // d_in[1]=x_r, d_in[2]=y : unused by the reference
  const float* Wi   = (const float*)d_in[3];
  const float* bi   = (const float*)d_in[4];
  const float* Ws   = (const float*)d_in[5];
  const float* bs   = (const float*)d_in[6];
  const float* temp = (const float*)d_in[7];
  const float* w_re = (const float*)d_in[8];
  const float* w_im = (const float*)d_in[9];
  const float* Wo   = (const float*)d_in[10];
  const float* bo   = (const float*)d_in[11];
  float* out = (float*)d_out;

  char* p = (char*)d_ws;
  auto alloc = [&](size_t bytes) -> char* {
    char* q = p;
    p += (bytes + 255) & ~(size_t)255;
    return q;
  };
  __bf16* xb     = (__bf16*)alloc(16384ull * 1024 * 2);   // x_q bf16
  __bf16* wib    = (__bf16*)alloc(1024ull * 1024 * 2);    // Wi bf16 [n][k]
  __bf16* wsb    = (__bf16*)alloc(256ull * 64 * 2);       // Ws bf16 [s][d]
  __bf16* wob    = (__bf16*)alloc(1024ull * 1024 * 2);    // Wo bf16 [n][k]
  __bf16* tmpb   = (__bf16*)alloc(16384ull * 1024 * 2);   // tmp [b,h,t,d]
  float*  logits = (float*)alloc(64ull * 4096 * 256 * 4); // [bh][t][s]
  float*  rmax   = (float*)alloc(64ull * 4096 * 4);
  float*  rsum   = (float*)alloc(64ull * 4096 * 4);
  float*  cmax   = (float*)alloc(64ull * 256 * 4);
  float*  csum   = (float*)alloc(64ull * 256 * 4);
  float*  zbuf   = (float*)alloc(64ull * 256 * 64 * 4);   // [bh][s][d]
  __bf16* outT   = (__bf16*)alloc(64ull * 64 * 256 * 2);  // [bh][d][s]
  __bf16* y2     = (__bf16*)alloc(16384ull * 1024 * 2);   // [b,t,c] bf16

  f2bf_kernel<<<(16384 * 1024 + 255) / 256, 256, 0, stream>>>(x_q, xb, 16384 * 1024);
  f2bf_kernel<<<(1024 * 1024 + 255) / 256, 256, 0, stream>>>(Wi, wib, 1024 * 1024);
  f2bf_kernel<<<(256 * 64 + 255) / 256, 256, 0, stream>>>(Ws, wsb, 256 * 64);
  f2bf_kernel<<<(1024 * 1024 + 255) / 256, 256, 0, stream>>>(Wo, wob, 1024 * 1024);

  // 1. in_proj (+bias) -> tmp[b,h,t,d] bf16
  gemm_big_kernel<<<dim3(128, 16), 256, 0, stream>>>(
      xb, wib, bi, nullptr, tmpb, 1024, 1024, 1024, 1024, 0);
  // 2. logits = (tmp @ Ws^T + bs) / temp
  gemm_logits_kernel<<<dim3(32, 4, 64), 256, 0, stream>>>(tmpb, wsb, bs, temp, logits);
  // 3. softmax stats over S (rows) and over T (columns)
  rowstats_kernel<<<64 * 4096 / 8, 256, 0, stream>>>(logits, rmax, rsum);
  colstats_kernel<<<64, 256, 0, stream>>>(logits, cmax, csum);
  // 4. slice-attention pooling z = softmax_T(logits)^T @ tmp
  gemm_pool_kernel<<<dim3(2, 1, 64), 256, 0, stream>>>(logits, tmpb, cmax, csum, zbuf);
  // 5. truncated rfft + mode mixing + irfft
  fourier_kernel<<<64, 256, 0, stream>>>(zbuf, w_re, w_im, outT);
  // 6. scatter back to tokens: softmax_S(logits) @ outPool
  gemm_scatter_kernel<<<dim3(32, 1, 64), 256, 0, stream>>>(logits, outT, rmax, rsum, y2);
  // 7. out_proj (+bias) -> fp32 output
  gemm_big_kernel<<<dim3(128, 16), 256, 0, stream>>>(
      y2, wob, bo, out, nullptr, 1024, 1024, 1024, 1024, 1);
}
